// SpatialCrossAttention_88751204204842
// MI455X (gfx1250) — matrix operand order repaired and checked by reading, not compiled, not run on hardware
//
#include <hip/hip_runtime.h>
#include <hip/hip_bf16.h>
#include <math.h>

// SpatialCrossAttention for MI455X (gfx1250), wave32 + bf16 WMMA.
//
// Algebraic collapse: q_map is spatially constant -> sim is i-invariant;
// conv1d pooling commutes with w_out ->
//   A[b,h,j] = sum_i w1d[i] * softmax_j(sim_base[b,h,j] + cpb_bias[b,i,j,h])
//   Z[b,c]   = sum_j v[b,c,j] * A[b,c/16,j]
//   y[b,:]   = LN( w_out @ Z[b] + b_out*Sum(w1d) + c1b + q[b] )
//
// All matrix ops via v_wmma_f32_16x16x32_bf16; operands pre-packed as bf16 so
// each WMMA operand half is a single 16B (b128) load. Writebacks keep uniform
// branches scalar (no per-lane exec churn beyond the inherent n<18 scatter).

#define DIM    256
#define HEADS  16
#define DHEAD  16
#define HH     12
#define WW     26
#define HW     312
#define HK     3
#define WK     6
#define P      18          // kv positions
#define SCALEQ 0.25f       // DHEAD^-0.5
#define OSCALE 4.0f
#define EPSLN  1e-5f
#define GSPLIT 4

// workspace layout (float offsets); total 1,638,400 floats (~6.6 MB)
#define S_OFF    0          // [256][18] depthwise border sums
#define SW_OFF   4608       // scalar sum(conv1d_w)
#define G_OFF    8192       // [256][18][2] normalized kv grid
#define SIM_OFF  20480      // [256][16][18] sim_base
#define A_OFF    98304      // [256][16][18] attention accumulator
#define Z_OFF    180224     // (spare)
#define Y_OFF    245760     // [256][256]
#define V_OFF    327680     // [256][256][18] v projection (f32)
#define WKB_OFF  1507328    // bf16 wk   [256*256] ushort
#define WVB_OFF  1540096    // bf16 wv
#define WOB_OFF  1572864    // bf16 w_out
#define ZB_OFF   1605632    // bf16 Z [256*256] ushort

typedef __attribute__((ext_vector_type(16))) __bf16 v16bf;
typedef __attribute__((ext_vector_type(8)))  float  v8f;

static __device__ inline unsigned short f2bfbits(float f) {
  union { float f; unsigned u; } v; v.f = f;
  return (unsigned short)((v.u + 0x7FFFu + ((v.u >> 16) & 1u)) >> 16);  // RNE
}
union V16 { unsigned short s[16]; uint4 q[2]; v16bf v; };
// build a v16bf from two 16B runs of bf16 (A halves or B halves)
static __device__ inline v16bf ldbf16x2(const unsigned short* p0,
                                        const unsigned short* p1) {
  V16 t;
  t.q[0] = *(const uint4*)p0;
  t.q[1] = *(const uint4*)p1;
  return t.v;
}
// 16-bit A-matrix 16x32 lane->K mapping (CDNA5 ISA 7.12.2):
// lanes 0-15 hold K {0..7,16..23}; lanes 16-31 hold K {8..15,24..31}.
// => per lane, elements 0..7 start at K = (lane<16?0:8), elements 8..15 at +16.

// ---------------------------------------------------------------- kernel 0
__global__ void k0_prep(const float* __restrict__ w_dw,
                        const float* __restrict__ w1d,
                        float* __restrict__ ws) {
  int c = threadIdx.x;
  for (int i = 0; i < HK; ++i)
    for (int j = 0; j < WK; ++j) {
      float s = 0.f;
      for (int kh = 0; kh < 6; ++kh) {
        int r = i * 4 - 1 + kh;
        if (r < 0 || r >= HH) continue;
        for (int kw = 0; kw < 6; ++kw) {
          int cc = j * 4 - 1 + kw;
          if (cc < 0 || cc >= WW) continue;
          s += w_dw[c * 36 + kh * 6 + kw];
        }
      }
      ws[S_OFF + c * P + i * WK + j] = s;
    }
  if (c == 0) {
    float sw = 0.f;
    for (int t = 0; t < HW; ++t) sw += w1d[t];
    ws[SW_OFF] = sw;
  }
}

// --------------------------------------------------- weight bf16 conversion
__global__ void kWb_conv(const float* __restrict__ wk,
                         const float* __restrict__ wv,
                         const float* __restrict__ w_out,
                         float* __restrict__ ws) {
  unsigned short* wkb = (unsigned short*)(ws + WKB_OFF);
  unsigned short* wvb = (unsigned short*)(ws + WVB_OFF);
  unsigned short* wob = (unsigned short*)(ws + WOB_OFF);
  int idx = blockIdx.x * 256 + threadIdx.x;   // grid = 768 blocks
  int m = idx >> 16, e = idx & 65535;
  if (m == 0)      wkb[e] = f2bfbits(wk[e]);
  else if (m == 1) wvb[e] = f2bfbits(wv[e]);
  else             wob[e] = f2bfbits(w_out[e]);
}

// ---------------------------------------------------------------- kernel A
__global__ void __launch_bounds__(256) kA_batch(
    const float* __restrict__ x, const float* __restrict__ q,
    const float* __restrict__ b_dw, const float* __restrict__ w_pw,
    float* __restrict__ ws) {
  __shared__ float sT[DIM * P];             // gelu(t); reused later for k result
  __shared__ unsigned short sKVb[32 * DIM]; // bf16 kv, [p][c], rows >=18 zeroed
  __shared__ float sOff[2 * P];
  __shared__ int   sIX[P], sIY[P];
  __shared__ float sFX[P], sFY[P];

  const int b    = blockIdx.x;
  const int tid  = threadIdx.x;
  const int lane = tid & 31;
  const int wid  = tid >> 5;
  const unsigned short* wkb = (const unsigned short*)(ws + WKB_OFF);
  const unsigned short* wvb = (const unsigned short*)(ws + WVB_OFF);

  // 1. depthwise conv on constant map == q_c * S + b ; exact GELU
  {
    float qc = q[b * DIM + tid];
    float bb = b_dw[tid];
    for (int p = 0; p < P; ++p) {
      float t = fmaf(qc, ws[S_OFF + tid * P + p], bb);
      sT[tid * P + p] = 0.5f * t * (1.f + erff(t * 0.70710678f));
    }
  }
  __syncthreads();
  // 2. pointwise 1x1 conv -> tanh -> offsets
  if (tid < 2 * P) {
    int o = tid / P, p = tid % P;
    float acc = 0.f;
    for (int c2 = 0; c2 < DIM; ++c2)
      acc = fmaf(w_pw[o * DIM + c2], sT[c2 * P + p], acc);
    sOff[o * P + p] = tanhf(acc) * OSCALE;
  }
  __syncthreads();
  // 3. kv grid (faithful quirky normalization) + bilinear coords
  if (tid < P) {
    int i = tid / WK, j = tid % WK;
    float vg0 = (float)j + sOff[tid];          // x channel + off0
    float vg1 = (float)i + sOff[P + tid];      // y channel + off1
    float n0 = vg0 - 1.f;                      // 2*vg0/(HK-1)-1, HK-1==2
    float n1 = 2.f * vg1 / 5.f - 1.f;          // 2*vg1/(WK-1)-1
    ws[G_OFF + b * 36 + tid * 2 + 0] = n0;
    ws[G_OFF + b * 36 + tid * 2 + 1] = n1;
    float xs = ((n0 + 1.f) * (float)WW - 1.f) * 0.5f;
    float ys = ((n1 + 1.f) * (float)HH - 1.f) * 0.5f;
    float x0 = floorf(xs), y0 = floorf(ys);
    sIX[tid] = (int)x0; sIY[tid] = (int)y0;
    sFX[tid] = xs - x0; sFY[tid] = ys - y0;
  }
  __syncthreads();
  // 4. bilinear grid-sample (zeros padding) -> bf16 staging, zero pad rows
  {
    const float* xb = x + (size_t)b * DIM * HW + (size_t)tid * HW;
    for (int p = 0; p < P; ++p) {
      int x0 = sIX[p], y0 = sIY[p];
      float fx = sFX[p], fy = sFY[p];
      float v = 0.f;
#pragma unroll
      for (int dy = 0; dy < 2; ++dy) {
        int yy = y0 + dy;
        if (yy < 0 || yy >= HH) continue;
        float wy = dy ? fy : (1.f - fy);
#pragma unroll
        for (int dx = 0; dx < 2; ++dx) {
          int xx = x0 + dx;
          if (xx < 0 || xx >= WW) continue;
          v = fmaf(wy * (dx ? fx : (1.f - fx)), xb[yy * WW + xx], v);
        }
      }
      sKVb[p * DIM + tid] = f2bfbits(v);
    }
    for (int p = P; p < 32; ++p) sKVb[p * DIM + tid] = 0;
  }
  __syncthreads();
  // 5. k = wk@kv, v = wv@kv via bf16 WMMA; 32 (mat,otile) jobs over 8 waves.
  //    mat and otile are wave-uniform -> keep the k/v writeback branch scalar.
  {
    const int koffA = (lane < 16) ? 0 : 8;    // A halves: K0 = koffA, K1 = koffA+16
    const int koffB = (lane < 16) ? 0 : 8;    // B halves: K0 = koffB, K1 = koffB+16
    for (int jb = wid; jb < 32; jb += 8) {
      const int mat   = jb >> 4;               // 0:k 1:v (uniform per wave-iter)
      const int otile = jb & 15;
      const unsigned short* wrow =
          (mat ? wvb : wkb) + (otile * 16 + (lane & 15)) * DIM;
      for (int ntile = 0; ntile < 2; ++ntile) {
        const int n = ntile * 16 + (lane & 15);  // kv position (rows >=18 are 0)
        const unsigned short* brow = sKVb + n * DIM;
        v8f acc; for (int r = 0; r < 8; ++r) acc[r] = 0.f;
        for (int kb = 0; kb < 8; ++kb) {
          v16bf av = ldbf16x2(wrow + kb * 32 + koffA, wrow + kb * 32 + koffA + 16);
          v16bf bv = ldbf16x2(brow + kb * 32 + koffB, brow + kb * 32 + koffB + 16);
          acc = __builtin_amdgcn_wmma_f32_16x16x32_bf16(
              false, av, false, bv, (short)0, acc, false, false);
        }
        const int mrow = otile * 16 + ((lane >> 4) << 3);
        const bool nok = (n < P);
        if (mat == 0) {                        // scalar branch (uniform)
#pragma unroll
          for (int r = 0; r < 8; ++r)
            if (nok) sT[(mrow + r) * P + n] = acc[r];
        } else {
          float* vdst = ws + V_OFF + (size_t)b * 4608 + n;
#pragma unroll
          for (int r = 0; r < 8; ++r)
            if (nok) vdst[(mrow + r) * P] = acc[r];
        }
      }
    }
  }
  __syncthreads();
  // 6. sim_base[h][j] (i-invariant!) ; 7. zero attention accumulator
  for (int e = tid; e < HEADS * P; e += 256) {
    int h = e / P, j = e % P;
    float s = 0.f;
    for (int d = 0; d < DHEAD; ++d)
      s = fmaf(q[b * DIM + h * DHEAD + d], sT[(h * DHEAD + d) * P + j], s);
    ws[SIM_OFF + b * 288 + e] = s * SCALEQ;
    ws[A_OFF + b * 288 + e] = 0.f;
  }
}

// ---------------------------------------------------------------- kernel B
__global__ void __launch_bounds__(256) kB_cpb(
    const float* __restrict__ w1, const float* __restrict__ b1,
    const float* __restrict__ w2, const float* __restrict__ b2,
    const float* __restrict__ w3, const float* __restrict__ b3,
    const float* __restrict__ w1d, float* __restrict__ ws) {
  __shared__ float sW1[128], sB1[64], sB2[64], sB3[16];
  __shared__ unsigned short sW2b[64 * 64];     // bf16 w2
  __shared__ unsigned short sW3b[16 * 64];     // bf16 w3
  __shared__ float sSim[288], sGrid[36];
  __shared__ unsigned short sH2[8][32 * 64];   // per-wave h2, bf16
  __shared__ float sSm[8][32 * 16];            // per-wave sim rows

  const int b    = blockIdx.x / GSPLIT;
  const int seg  = blockIdx.x % GSPLIT;
  const int tid  = threadIdx.x;
  const int lane = tid & 31, wid = tid >> 5;

  for (int e = tid; e < 128;  e += 256) sW1[e] = w1[e];
  for (int e = tid; e < 64;   e += 256) { sB1[e] = b1[e]; sB2[e] = b2[e]; }
  for (int e = tid; e < 4096; e += 256) sW2b[e] = f2bfbits(w2[e]);
  for (int e = tid; e < 1024; e += 256) sW3b[e] = f2bfbits(w3[e]);
  for (int e = tid; e < 16;   e += 256) sB3[e] = b3[e];
  for (int e = tid; e < 288;  e += 256) sSim[e] = ws[SIM_OFF + b * 288 + e];
  for (int e = tid; e < 36;   e += 256) sGrid[e] = ws[G_OFF + b * 36 + e];
  __syncthreads();

  float aacc[P];
#pragma unroll
  for (int j = 0; j < P; ++j) aacc[j] = 0.f;

  const int koffA = (lane < 16) ? 0 : 8;
  const int koffB = (lane < 16) ? 0 : 8;
  const int gw    = seg * 8 + wid;
  for (int i = gw; i < HW; i += 32) {
    float gq0 = 2.f * (float)(i % WW) / 11.f - 1.f;   // faithful: / (H-1)
    float gq1 = 2.f * (float)(i / WW) / 25.f - 1.f;   // faithful: / (W-1)
    // ---- layer1 (VALU, built directly in A layout) + layer2 (WMMA)
    for (int mt = 0; mt < 2; ++mt) {
      int j = mt * 16 + (lane & 15);
      float u0 = 0.f, u1 = 0.f;
      if (j < P) {
        float p0 = gq0 - sGrid[j * 2 + 0];
        float p1 = gq1 - sGrid[j * 2 + 1];
        u0 = copysignf(log1pf(fabsf(p0)), p0);
        u1 = copysignf(log1pf(fabsf(p1)), p1);
      }
      v16bf a2[2];
#pragma unroll
      for (int kb = 0; kb < 2; ++kb) {
        V16 t;
#pragma unroll
        for (int e = 0; e < 16; ++e) {
          int k = kb * 32 + koffA + (e < 8 ? e : e + 8);
          float h1 = fmaf(sW1[2 * k], u0, fmaf(sW1[2 * k + 1], u1, sB1[k]));
          t.s[e] = f2bfbits(fmaxf(h1, 0.f));
        }
        a2[kb] = t.v;
      }
      for (int nt = 0; nt < 4; ++nt) {
        const int n = nt * 16 + (lane & 15);
        const unsigned short* brow = sW2b + n * 64;
        v8f acc; for (int r = 0; r < 8; ++r) acc[r] = 0.f;
#pragma unroll
        for (int kb = 0; kb < 2; ++kb) {
          v16bf bv = ldbf16x2(brow + kb * 32 + koffB, brow + kb * 32 + koffB + 16);
          acc = __builtin_amdgcn_wmma_f32_16x16x32_bf16(
              false, a2[kb], false, bv, (short)0, acc, false, false);
        }
        unsigned short* h2dst = &sH2[wid][(mt * 16 + ((lane >> 4) << 3)) * 64 + n];
#pragma unroll
        for (int r = 0; r < 8; ++r)
          h2dst[r * 64] = f2bfbits(fmaxf(acc[r] + sB2[n], 0.f));
      }
    }
    // ---- layer3 (WMMA, N = 16 heads)
    for (int mt = 0; mt < 2; ++mt) {
      const unsigned short* arow = &sH2[wid][(mt * 16 + (lane & 15)) * 64];
      const unsigned short* brow = sW3b + (lane & 15) * 64;
      v8f acc; for (int r = 0; r < 8; ++r) acc[r] = 0.f;
#pragma unroll
      for (int kb = 0; kb < 2; ++kb) {
        v16bf a3 = ldbf16x2(arow + kb * 32 + koffA, arow + kb * 32 + koffA + 16);
        v16bf bv = ldbf16x2(brow + kb * 32 + koffB, brow + kb * 32 + koffB + 16);
        acc = __builtin_amdgcn_wmma_f32_16x16x32_bf16(
            false, a3, false, bv, (short)0, acc, false, false);
      }
      int h = lane & 15;
      int jbase = mt * 16 + ((lane >> 4) << 3);
#pragma unroll
      for (int r = 0; r < 8; ++r) {
        int j = jbase + r;
        if (j < P) sSm[wid][j * 16 + h] = sSim[h * P + j] + acc[r] + sB3[h];
      }
    }
    // ---- per-head softmax over 18 kv rows, pooled-accumulate
    float w1di = w1d[i];
    if (lane < 16) {
      int h = lane;
      float sv[P], m = -1e30f;
#pragma unroll
      for (int j = 0; j < P; ++j) { sv[j] = sSm[wid][j * 16 + h]; m = fmaxf(m, sv[j]); }
      float s = 0.f;
#pragma unroll
      for (int j = 0; j < P; ++j) { sv[j] = expf(sv[j] - m); s += sv[j]; }
      float inv = w1di / s;
#pragma unroll
      for (int j = 0; j < P; ++j) aacc[j] = fmaf(sv[j], inv, aacc[j]);
    }
  }
  if (lane < 16)
    for (int j = 0; j < P; ++j)
      atomicAdd(&ws[A_OFF + b * 288 + lane * P + j], aacc[j]);
}

// ---------------------------------------------------------------- kernel C1
__global__ void kC1_z(float* __restrict__ ws) {
  int b = blockIdx.x, c = threadIdx.x, h = c >> 4;
  float z = 0.f;
  for (int j = 0; j < P; ++j)
    z = fmaf(ws[V_OFF + (size_t)b * 4608 + c * P + j],
             ws[A_OFF + b * 288 + h * P + j], z);
  ((unsigned short*)(ws + ZB_OFF))[b * DIM + c] = f2bfbits(z);
}

// ---------------------------------------------------------------- kernel C2
__global__ void __launch_bounds__(256) kC2_out(
    const float* __restrict__ b_out, const float* __restrict__ c1b,
    const float* __restrict__ q, float* __restrict__ ws) {
  const int tid = threadIdx.x, lane = tid & 31, wid = tid >> 5;
  const int tile  = blockIdx.x * 8 + wid;
  const int otile = tile >> 4, btile = tile & 15;
  const float sw = ws[SW_OFF], cb = c1b[0];
  const unsigned short* wob = (const unsigned short*)(ws + WOB_OFF);
  const unsigned short* zb  = (const unsigned short*)(ws + ZB_OFF);
  const int koffA = (lane < 16) ? 0 : 8;
  const int koffB = (lane < 16) ? 0 : 8;
  const unsigned short* arow = wob + (otile * 16 + (lane & 15)) * DIM;
  const unsigned short* brow = zb + (btile * 16 + (lane & 15)) * DIM;
  v8f acc; for (int r = 0; r < 8; ++r) acc[r] = 0.f;
  for (int kb = 0; kb < 8; ++kb) {
    v16bf av = ldbf16x2(arow + kb * 32 + koffA, arow + kb * 32 + koffA + 16);
    v16bf bv = ldbf16x2(brow + kb * 32 + koffB, brow + kb * 32 + koffB + 16);
    acc = __builtin_amdgcn_wmma_f32_16x16x32_bf16(
        false, av, false, bv, (short)0, acc, false, false);
  }
  const int bcol = btile * 16 + (lane & 15);
  const int obase = otile * 16 + ((lane >> 4) << 3);
#pragma unroll
  for (int r = 0; r < 8; ++r) {
    int o = obase + r;
    ws[Y_OFF + bcol * DIM + o] = acc[r] + b_out[o] * sw + cb + q[bcol * DIM + o];
  }
}

// ---------------------------------------------------------------- kernel D
__global__ void __launch_bounds__(256) kD_ln(
    const float* __restrict__ g, const float* __restrict__ be,
    const float* __restrict__ ws, float* __restrict__ out) {
  __shared__ float red[256];
  int b = blockIdx.x, c = threadIdx.x;
  float y = ws[Y_OFF + b * DIM + c];
  red[c] = y; __syncthreads();
  for (int s = 128; s > 0; s >>= 1) { if (c < s) red[c] += red[c + s]; __syncthreads(); }
  float mu = red[0] / (float)DIM; __syncthreads();
  float d = y - mu;
  red[c] = d * d; __syncthreads();
  for (int s = 128; s > 0; s >>= 1) { if (c < s) red[c] += red[c + s]; __syncthreads(); }
  float var = red[0] / (float)DIM;
  out[b * DIM + c] = d * rsqrtf(var + EPSLN) * g[c] + be[c];
}

// ---------------------------------------------------------------- launch
extern "C" void kernel_launch(void* const* d_in, const int* in_sizes, int n_in,
                              void* d_out, int out_size, void* d_ws, size_t ws_size,
                              hipStream_t stream) {
  (void)in_sizes; (void)n_in; (void)out_size; (void)ws_size;
  const float* x     = (const float*)d_in[0];
  const float* q     = (const float*)d_in[1];
  const float* w_dw  = (const float*)d_in[2];
  const float* b_dw  = (const float*)d_in[3];
  const float* w_pw  = (const float*)d_in[4];
  const float* wk    = (const float*)d_in[5];
  const float* wv    = (const float*)d_in[6];
  const float* w_out = (const float*)d_in[7];
  const float* b_out = (const float*)d_in[8];
  const float* cw1   = (const float*)d_in[9];
  const float* cb1   = (const float*)d_in[10];
  const float* cw2   = (const float*)d_in[11];
  const float* cb2   = (const float*)d_in[12];
  const float* cw3   = (const float*)d_in[13];
  const float* cb3   = (const float*)d_in[14];
  const float* w1d   = (const float*)d_in[15];
  const float* c1b   = (const float*)d_in[16];
  const float* lng   = (const float*)d_in[17];
  const float* lnb   = (const float*)d_in[18];
  float* ws  = (float*)d_ws;   // needs ~6.6 MB
  float* out = (float*)d_out;  // (256,1,256) f32

  k0_prep <<<1,            256, 0, stream>>>(w_dw, w1d, ws);
  kWb_conv<<<768,          256, 0, stream>>>(wk, wv, w_out, ws);
  kA_batch<<<256,          256, 0, stream>>>(x, q, b_dw, w_pw, ws);
  kB_cpb  <<<256 * GSPLIT, 256, 0, stream>>>(cw1, cb1, cw2, cb2, cw3, cb3, w1d, ws);
  kC1_z   <<<256,          256, 0, stream>>>(ws);
  kC2_out <<<32,           256, 0, stream>>>(b_out, c1b, q, ws);
  kD_ln   <<<256,          256, 0, stream>>>(lng, lnb, ws, out);
}
